// RWKVAttention_56315611185630
// MI455X (gfx1250) — matrix operand order repaired and checked by reading, not compiled, not run on hardware
//
#include <hip/hip_runtime.h>
#include <math.h>

#define B_  4
#define T_  2048
#define C_  1024
#define H_  16
#define D_  64
#define EPSL 1e-5f

typedef __attribute__((ext_vector_type(16))) _Float16 v16h;
typedef __attribute__((ext_vector_type(8)))  _Float16 v8h;
typedef __attribute__((ext_vector_type(8)))  float    v8f;
typedef __attribute__((ext_vector_type(4)))  unsigned int u32x4;
typedef __attribute__((ext_vector_type(8)))  int          i32x8;
typedef __attribute__((ext_vector_type(4)))  int          i32x4;

union Frag { v16h v; v8h h[2]; };

// ---------------- TDM: DMA a [rows=128] x [k=32] f16 tile (row stride = K) into LDS.
// D# built per CDNA5 ISA 8.3/8.4: group0 = {count|flags, lds_addr, global_addr, type=2},
// group1 = {data_size, tensor_dim0/1, tile_dim0/1, tensor_dim0_stride}. Groups 2/3 zero
// (2D tile: tile_dim2/3/4 = 0 = unused). workgroup_mask = 0 (not in a cluster).
__device__ inline void tdm_load_b_tile(unsigned ldsAddr, unsigned long long gaddr,
                                       int tensorK, int tensorN)
{
    u32x4 g0; i32x8 g1; i32x4 g2 = {0, 0, 0, 0}; i32x4 g3 = {0, 0, 0, 0};
    const unsigned d0 = (unsigned)tensorK;    // tensor_dim0 (elements along k)
    const unsigned d1 = (unsigned)tensorN;    // tensor_dim1 (rows)
    g0[0] = 1u;                                            // count=1 user descriptor
    g0[1] = ldsAddr;                                       // LDS byte offset
    g0[2] = (unsigned)(gaddr & 0xFFFFFFFFull);             // global_addr[31:0]
    g0[3] = (unsigned)((gaddr >> 32) & 0x01FFFFFFull)      // global_addr[56:32]
          | (2u << 30);                                    // type=2 ("image")
    g1[0] = (int)(1u << 16);                               // data_size=1 -> 2-byte elems
    g1[1] = (int)((d0 & 0xFFFFu) << 16);                   // tensor_dim0[15:0]
    g1[2] = (int)(((d0 >> 16) & 0xFFFFu) | ((d1 & 0xFFFFu) << 16)); // dim0 hi | dim1 lo
    g1[3] = (int)(((d1 >> 16) & 0xFFFFu) | (32u << 16));   // dim1 hi | tile_dim0=32
    g1[4] = 128;                                           // tile_dim1=128, tile_dim2=0
    g1[5] = (int)d0;                                       // tensor_dim0_stride = K
    g1[6] = 0;                                             // stride0 hi | stride1 lo
    g1[7] = 0;
#if __has_include(<hip/amd_detail/amd_gfx1250_TDM.h>)
    i32x8 g4 = {0, 0, 0, 0, 0, 0, 0, 0};
    __builtin_amdgcn_tensor_load_to_lds(g0, g1, g2, g3, g4, 0);
#else
    __builtin_amdgcn_tensor_load_to_lds(g0, g1, g2, g3, 0);
#endif
}

// ---------------- prep: token shift + time-mix -> 4 f16 activation arrays ----
__global__ __launch_bounds__(256)
void prep_kernel(const float* __restrict__ x,
                 const float* __restrict__ mk, const float* __restrict__ mv,
                 const float* __restrict__ mr, const float* __restrict__ mg,
                 void* xk_, void* xv_, void* xr_, void* xg_)
{
    int idx = blockIdx.x * blockDim.x + threadIdx.x;      // [0, B*T*C)
    int c = idx % C_;
    int t = (idx / C_) % T_;
    int b = idx / (C_ * T_);
    int tprev = (t == 0) ? 1 : (t - 1);
    float cur  = x[idx];
    float prev = x[((size_t)b * T_ + tprev) * C_ + c];
    float xx = prev - cur;
    ((_Float16*)xk_)[idx] = (_Float16)(cur + xx * mk[c]);
    ((_Float16*)xv_)[idx] = (_Float16)(cur + xx * mv[c]);
    ((_Float16*)xr_)[idx] = (_Float16)(cur + xx * mr[c]);
    ((_Float16*)xg_)[idx] = (_Float16)(cur + xx * mg[c]);
}

// ---------------- f32 -> f16 weight convert ----------------
__global__ __launch_bounds__(256)
void cvt_kernel(const float* __restrict__ in, void* out_)
{
    int idx = blockIdx.x * blockDim.x + threadIdx.x;
    ((_Float16*)out_)[idx] = (_Float16)in[idx];
}

// ---------------- WMMA GEMM: out[m,n] = sum_k A[m,k] * W[n,k] ----------------
// Block tile 128x128: 8 waves as 2(M) x 4(N), each wave 64x32 (8 WMMA accumulators).
// B tile (128 rows x 32 k) staged in LDS by the Tensor Data Mover, double-buffered.
// MODE 0: store f32.  MODE 1: store sigmoid() as f16 (gate path).
template<int MODE>
__global__ __launch_bounds__(256)
void gemm_kernel(const void* __restrict__ A_, const void* __restrict__ W_,
                 void* __restrict__ out_, int M, int N, int K)
{
    __shared__ __align__(16) _Float16 Bsm[2][128 * 32];   // 2 x 8 KB

    const _Float16* A = (const _Float16*)A_;
    const _Float16* W = (const _Float16*)W_;
    const int lane  = threadIdx.x & 31;
    const int wave  = threadIdx.x >> 5;
    const int wm    = wave & 1;              // 2 waves along M
    const int wn    = wave >> 1;             // 4 waves along N
    const int mBase = blockIdx.y * 128 + wm * 64;
    const int nBlk  = blockIdx.x * 128;
    const int nLoc  = wn * 32;               // wave's N offset inside the block tile
    const int lhalf = lane >> 4;             // 0 or 1 (lane half)
    const int l16   = lane & 15;

    v8f acc[4][2];
    #pragma unroll
    for (int ti = 0; ti < 4; ++ti)
        #pragma unroll
        for (int tn = 0; tn < 2; ++tn) { v8f z = {}; acc[ti][tn] = z; }

    const _Float16* aRow[4];
    #pragma unroll
    for (int ti = 0; ti < 4; ++ti)
        aRow[ti] = A + (size_t)(mBase + ti * 16 + l16) * K;

    const unsigned ldsBase   = (unsigned)(uintptr_t)&Bsm[0][0]; // low 32b = LDS offset
    const unsigned ldsStride = 128u * 32u * 2u;                 // 8192 bytes
    const unsigned long long gW =
        (unsigned long long)(uintptr_t)(W + (size_t)nBlk * K);  // byte addr of W[nBlk,0]

    const int steps = K / 32;
    if (threadIdx.x < 32)                    // wave 0 drives the TDM
        tdm_load_b_tile(ldsBase, gW, K, N);

    #pragma unroll 2
    for (int i = 0; i < steps; ++i) {
        const int kk = i * 32;
        if (threadIdx.x < 32)
            __builtin_amdgcn_s_wait_tensorcnt((short)0);   // current tile landed in LDS
        __syncthreads();                                   // publish to all 8 waves
        if (threadIdx.x < 32 && (i + 1) < steps)           // prefetch next tile via DMA
            tdm_load_b_tile(ldsBase + (unsigned)((i + 1) & 1) * ldsStride,
                            gW + (unsigned long long)(kk + 32) * 2ull, K, N);

        // A fragments: 16x32 f16; lane half 0 holds K {0..7,16..23}, half 1 {8..15,24..31}
        Frag a[4], bf[2];
        const int ka = kk + lhalf * 8;
        #pragma unroll
        for (int ti = 0; ti < 4; ++ti) {
            a[ti].h[0] = *(const v8h*)(aRow[ti] + ka);
            a[ti].h[1] = *(const v8h*)(aRow[ti] + ka + 16);
        }
        // B fragments from LDS tile: lane half 0 holds K {0..7,16..23}, half 1 {8..15,24..31}
        const _Float16* Bl = &Bsm[i & 1][0];
        #pragma unroll
        for (int tn = 0; tn < 2; ++tn) {
            const _Float16* rowp = Bl + (size_t)(nLoc + tn * 16 + l16) * 32 + lhalf * 8;
            bf[tn].h[0] = *(const v8h*)(rowp);
            bf[tn].h[1] = *(const v8h*)(rowp + 16);
        }

        #pragma unroll
        for (int ti = 0; ti < 4; ++ti) {
            acc[ti][0] = __builtin_amdgcn_wmma_f32_16x16x32_f16(false, a[ti].v, false, bf[0].v,
                                                                (short)0, acc[ti][0], false, false);
            acc[ti][1] = __builtin_amdgcn_wmma_f32_16x16x32_f16(false, a[ti].v, false, bf[1].v,
                                                                (short)0, acc[ti][1], false, false);
        }
        __syncthreads();                     // done reading this buffer before reuse
    }

    // Epilogue. C/D layout: lane<16 -> N=lane, M=e; lane>=16 -> N=lane-16, M=8+e.
    #pragma unroll
    for (int ti = 0; ti < 4; ++ti) {
        #pragma unroll
        for (int tn = 0; tn < 2; ++tn) {
            const int col  = nBlk + nLoc + tn * 16 + l16;
            const int rowB = mBase + ti * 16 + lhalf * 8;
            #pragma unroll
            for (int e = 0; e < 8; ++e) {
                float vv = acc[ti][tn][e];
                size_t o = (size_t)(rowB + e) * N + col;
                if (MODE == 0) {
                    ((float*)out_)[o] = vv;
                } else {
                    ((_Float16*)out_)[o] = (_Float16)(1.0f / (1.0f + __expf(-vv)));
                }
            }
        }
    }
}

// ---------------- RWKV scan: one block per (b,h), state in registers --------
// thread tid: i = tid>>2 (state row / k-dim), jg = tid&3 (16-wide j slice)
__global__ __launch_bounds__(256)
void scan_kernel(const float* __restrict__ R, const float* __restrict__ Kx,
                 const float* __restrict__ V, const float* __restrict__ td,
                 float* __restrict__ attn)
{
    const int bh = blockIdx.x;               // 0..63
    const int b  = bh / H_, h = bh % H_;
    const int i  = threadIdx.x >> 2;
    const int jg = threadIdx.x & 3;
    const float wd = __expf(-__expf(td[h * D_ + i]));   // decay for k-dim i

    float st[16];
    #pragma unroll
    for (int j = 0; j < 16; ++j) st[j] = 0.0f;

    size_t base = ((size_t)b * T_) * C_ + (size_t)h * D_;
    for (int t = 0; t < T_; ++t, base += C_) {
        const float ki = Kx[base + i];
        const float4* vp = (const float4*)(V + base + jg * 16);
        const float4* rp = (const float4*)(R + base + jg * 16);
        float4 v4[4] = {vp[0], vp[1], vp[2], vp[3]};
        float4 r4[4] = {rp[0], rp[1], rp[2], rp[3]};
        float acc = 0.0f;
        #pragma unroll
        for (int q = 0; q < 4; ++q) {
            st[q*4+0] = fmaf(st[q*4+0], wd, ki * v4[q].x); acc = fmaf(st[q*4+0], r4[q].x, acc);
            st[q*4+1] = fmaf(st[q*4+1], wd, ki * v4[q].y); acc = fmaf(st[q*4+1], r4[q].y, acc);
            st[q*4+2] = fmaf(st[q*4+2], wd, ki * v4[q].z); acc = fmaf(st[q*4+2], r4[q].z, acc);
            st[q*4+3] = fmaf(st[q*4+3], wd, ki * v4[q].w); acc = fmaf(st[q*4+3], r4[q].w, acc);
        }
        // reduce the 4 j-slices (adjacent lanes, same wave32)
        acc += __shfl_xor(acc, 1, 32);
        acc += __shfl_xor(acc, 2, 32);
        if (jg == 0) attn[base + i] = acc;
    }
}

// ---------------- gate multiply: a16 = f16(attn * g) ----------------
__global__ __launch_bounds__(256)
void mulg_kernel(const float* __restrict__ attn, const void* __restrict__ g16_,
                 void* __restrict__ a16_)
{
    int idx = blockIdx.x * blockDim.x + threadIdx.x;
    const _Float16* g = (const _Float16*)g16_;
    ((_Float16*)a16_)[idx] = (_Float16)(attn[idx] * (float)g[idx]);
}

// ---------------- LayerNorm over C=1024, one row per block ----------------
__global__ __launch_bounds__(256)
void ln_kernel(const float* __restrict__ y, const float* __restrict__ gamma,
               const float* __restrict__ beta, float* __restrict__ out)
{
    __shared__ float s1[256];
    __shared__ float s2[256];
    const int row = blockIdx.x;
    const int tid = threadIdx.x;
    float4 v = ((const float4*)(y + (size_t)row * C_))[tid];
    s1[tid] = v.x + v.y + v.z + v.w;
    s2[tid] = v.x*v.x + v.y*v.y + v.z*v.z + v.w*v.w;
    __syncthreads();
    for (int off = 128; off > 0; off >>= 1) {
        if (tid < off) { s1[tid] += s1[tid + off]; s2[tid] += s2[tid + off]; }
        __syncthreads();
    }
    const float mean = s1[0] * (1.0f / C_);
    const float var  = s2[0] * (1.0f / C_) - mean * mean;
    const float rstd = rsqrtf(var + EPSL);
    float4 g4 = ((const float4*)gamma)[tid];
    float4 b4 = ((const float4*)beta)[tid];
    float4 o;
    o.x = (v.x - mean) * rstd * g4.x + b4.x;
    o.y = (v.y - mean) * rstd * g4.y + b4.y;
    o.z = (v.z - mean) * rstd * g4.z + b4.z;
    o.w = (v.w - mean) * rstd * g4.w + b4.w;
    ((float4*)(out + (size_t)row * C_))[tid] = o;
}

extern "C" void kernel_launch(void* const* d_in, const int* in_sizes, int n_in,
                              void* d_out, int out_size, void* d_ws, size_t ws_size,
                              hipStream_t stream)
{
    (void)in_sizes; (void)n_in; (void)out_size; (void)ws_size;
    const float* x    = (const float*)d_in[0];
    const float* td   = (const float*)d_in[1];
    const float* mk   = (const float*)d_in[2];
    const float* mv   = (const float*)d_in[3];
    const float* mr   = (const float*)d_in[4];
    const float* mg   = (const float*)d_in[5];
    const float* Wr   = (const float*)d_in[6];
    const float* Wk   = (const float*)d_in[7];
    const float* Wv   = (const float*)d_in[8];
    const float* Wg   = (const float*)d_in[9];
    const float* Wo   = (const float*)d_in[10];
    const float* lng  = (const float*)d_in[11];
    const float* lnb  = (const float*)d_in[12];

    const size_t BTC = (size_t)B_ * T_ * C_;   // 8388608
    const size_t CC  = (size_t)C_ * C_;        // 1048576
    char* ws = (char*)d_ws;
    void* xk16 = ws + BTC * 0;                 // f16, 16 MB each
    void* xv16 = ws + BTC * 2;
    void* xr16 = ws + BTC * 4;
    void* xg16 = ws + BTC * 6;
    char* wbase = ws + BTC * 8;                // 5 f16 weights, 2 MB each
    void* Wr16 = wbase + CC * 0;
    void* Wk16 = wbase + CC * 2;
    void* Wv16 = wbase + CC * 4;
    void* Wg16 = wbase + CC * 6;
    void* Wo16 = wbase + CC * 8;
    char* fbase = wbase + CC * 10;
    float* r    = (float*)(fbase);             // f32, 32 MB
    float* k    = (float*)(fbase + BTC * 4);
    float* v    = (float*)(fbase + BTC * 8);
    void*  g16  = (void*) (fbase + BTC * 12);  // f16, 16 MB
    float* attn = (float*)(fbase + BTC * 14);  // f32, 32 MB
    void*  a16  = (void*) (fbase + BTC * 18);  // f16, 16 MB
    float* y    = (float*)(fbase + BTC * 20);  // f32, 32 MB

    const int M = B_ * T_, N = C_, K = C_;
    dim3 gemmGrid(N / 128, M / 128);           // (8, 64)
    dim3 blk(256);

    prep_kernel<<<dim3((unsigned)(BTC / 256)), blk, 0, stream>>>(x, mk, mv, mr, mg,
                                                                 xk16, xv16, xr16, xg16);
    cvt_kernel<<<dim3((unsigned)(CC / 256)), blk, 0, stream>>>(Wr, Wr16);
    cvt_kernel<<<dim3((unsigned)(CC / 256)), blk, 0, stream>>>(Wk, Wk16);
    cvt_kernel<<<dim3((unsigned)(CC / 256)), blk, 0, stream>>>(Wv, Wv16);
    cvt_kernel<<<dim3((unsigned)(CC / 256)), blk, 0, stream>>>(Wg, Wg16);
    cvt_kernel<<<dim3((unsigned)(CC / 256)), blk, 0, stream>>>(Wo, Wo16);

    gemm_kernel<0><<<gemmGrid, blk, 0, stream>>>(xr16, Wr16, r,   M, N, K);
    gemm_kernel<0><<<gemmGrid, blk, 0, stream>>>(xk16, Wk16, k,   M, N, K);
    gemm_kernel<0><<<gemmGrid, blk, 0, stream>>>(xv16, Wv16, v,   M, N, K);
    gemm_kernel<1><<<gemmGrid, blk, 0, stream>>>(xg16, Wg16, g16, M, N, K);

    scan_kernel<<<dim3(B_ * H_), blk, 0, stream>>>(r, k, v, td, attn);

    mulg_kernel<<<dim3((unsigned)(BTC / 256)), blk, 0, stream>>>(attn, g16, a16);

    gemm_kernel<0><<<gemmGrid, blk, 0, stream>>>(a16, Wo16, y, M, N, K);

    ln_kernel<<<dim3(B_ * T_), blk, 0, stream>>>(y, lng, lnb, (float*)d_out);
}
